// BNN_60782377173339
// MI455X (gfx1250) — matrix operand order repaired and checked by reading, not compiled, not run on hardware
//
#include <hip/hip_runtime.h>
#include <hip/hip_bf16.h>
#include <math.h>

typedef __attribute__((ext_vector_type(16))) _Float16 v16h;
typedef __attribute__((ext_vector_type(8)))  float    v8f;

#define BM 128
#define BN 64
#define BK 32
#define LSTR 40              // LDS row stride in f16 (32 + 8 pad; 80B = 5*16B, keeps b128 alignment)

union F16x16 { uint4 q[2]; v16h h; };
union F16x8  { uint4 q; _Float16 x[8]; };

// ---- CDNA5 async global->LDS (ASYNCcnt-tracked, no VGPR staging) -----------
__device__ __forceinline__ void async_ld_b128(uint32_t lds_off, const void* gptr)
{
    asm volatile("global_load_async_to_lds_b128 %0, %1, off"
                 :: "v"(lds_off), "v"((uint64_t)(uintptr_t)gptr)
                 : "memory");
}
__device__ __forceinline__ void wait_async0()
{
    asm volatile("s_wait_asynccnt 0x0" ::: "memory");
}

// ---------------------------------------------------------------------------
// Fused dual-GEMM:  mu = A @ W^T-layout,  var = (A.^2) @ V^T-layout
// A:  [M x K] f16 row-major (stride K).  Bw,Bv: [512 x K] f16 (N-major).
// EPI==0: write f32 mu/var (layer 1, M=2048).
// EPI==1: h = leaky_relu(mu + sqrt(var+1e-12)*eps), write f16 to hOut (stride 544).
// Double-buffered LDS fed by global_load_async_to_lds_b128; one barrier/K-step.
// ---------------------------------------------------------------------------
template<int EPI>
__global__ __launch_bounds__(256)
void gemm_fused(const _Float16* __restrict__ A,
                const _Float16* __restrict__ Bw,
                const _Float16* __restrict__ Bv,
                int K,
                const float* __restrict__ eps,
                float* __restrict__ muOut,
                float* __restrict__ varOut,
                _Float16* __restrict__ hOut)
{
    __shared__ _Float16 sA[2 * BM * LSTR];
    __shared__ _Float16 sW[2 * BN * LSTR];
    __shared__ _Float16 sV[2 * BN * LSTR];

    const int t    = threadIdx.x;
    const int lane = t & 31;
    const int wid  = t >> 5;
    const int wm   = wid & 3;        // 4 waves along M (32 rows each)
    const int wn   = wid >> 2;       // 2 waves along N (32 cols each)

    const int rowBase = blockIdx.x * BM;
    const int colBase = blockIdx.y * BN;

    v8f accMu[2][2] = {};
    v8f accVar[2][2] = {};

    // per-thread staging chunk: 16B of 8 f16
    const int ar0 = t >> 2;          // rows 0..63 (+64 for 2nd A chunk)
    const int ak  = (t & 3) * 8;     // k sub-offset within BK

    const int nk = K / BK;

    // global source pointers for this thread's chunks (advance by k0 each slab)
    const _Float16* gA0 = A  + (size_t)(rowBase + ar0)      * K + ak;
    const _Float16* gA1 = A  + (size_t)(rowBase + ar0 + 64) * K + ak;
    const _Float16* gW  = Bw + (size_t)(colBase + ar0)      * K + ak;
    const _Float16* gV  = Bv + (size_t)(colBase + ar0)      * K + ak;

    // LDS destination offsets (bytes; flat->LDS uses low 32 bits)
    const uint32_t lA0 = (uint32_t)(uintptr_t)&sA[(ar0)      * LSTR + ak];
    const uint32_t lA1 = (uint32_t)(uintptr_t)&sA[(ar0 + 64) * LSTR + ak];
    const uint32_t lW  = (uint32_t)(uintptr_t)&sW[(ar0)      * LSTR + ak];
    const uint32_t lV  = (uint32_t)(uintptr_t)&sV[(ar0)      * LSTR + ak];
    const uint32_t strideA = BM * LSTR * 2;   // byte distance between the 2 buffers
    const uint32_t strideB = BN * LSTR * 2;

    // fragment addressing (wave32 layouts per CDNA5 ISA 7.12.2)
    const int frow = lane & 15;
    const int asel = (lane >> 4) * 8;    // A: K chunks {asel..asel+7, 16+asel..16+asel+7}
    const int bsel = (lane >> 4) * 8;    // B: K chunks {bsel..bsel+7, 16+bsel..16+bsel+7}

    // prologue: async-load K-slab 0 into buffer 0
    async_ld_b128(lA0, gA0);
    async_ld_b128(lA1, gA1);
    async_ld_b128(lW,  gW);
    async_ld_b128(lV,  gV);

    for (int kk = 0; kk < nk; ++kk) {
        wait_async0();        // my 4 chunks of buf[kk&1] are in LDS
        __syncthreads();      // everyone's are; everyone done reading buf[(kk-1)&1]

        const int cur = kk & 1;
        if (kk + 1 < nk) {    // overlap next slab's DMA with this slab's WMMAs
            const int    nxt = (kk + 1) & 1;
            const size_t k0  = (size_t)(kk + 1) * BK;
            async_ld_b128(lA0 + nxt * strideA, gA0 + k0);
            async_ld_b128(lA1 + nxt * strideA, gA1 + k0);
            async_ld_b128(lW  + nxt * strideB, gW  + k0);
            async_ld_b128(lV  + nxt * strideB, gV  + k0);
        }

        const _Float16* bA = sA + cur * BM * LSTR;
        const _Float16* bW = sW + cur * BN * LSTR;
        const _Float16* bV = sV + cur * BN * LSTR;

        #pragma unroll
        for (int tm = 0; tm < 2; ++tm) {
            F16x16 a;
            const _Float16* pa = &bA[(wm * 32 + tm * 16 + frow) * LSTR];
            a.q[0] = *(const uint4*)(pa + asel);
            a.q[1] = *(const uint4*)(pa + 16 + asel);
            const v16h av  = a.h;
            const v16h asq = av * av;          // v_pk_mul_f16: square A in-register
            #pragma unroll
            for (int tn = 0; tn < 2; ++tn) {
                F16x16 w, v;
                const _Float16* pw = &bW[(wn * 32 + tn * 16 + frow) * LSTR + bsel];
                const _Float16* pv = &bV[(wn * 32 + tn * 16 + frow) * LSTR + bsel];
                w.q[0] = *(const uint4*)(pw);
                w.q[1] = *(const uint4*)(pw + 16);
                v.q[0] = *(const uint4*)(pv);
                v.q[1] = *(const uint4*)(pv + 16);
                accMu[tm][tn] = __builtin_amdgcn_wmma_f32_16x16x32_f16(
                    false, av,  false, w.h, (short)0, accMu[tm][tn], false, false);
                accVar[tm][tn] = __builtin_amdgcn_wmma_f32_16x16x32_f16(
                    false, asq, false, v.h, (short)0, accVar[tm][tn], false, false);
            }
        }
    }

    // epilogue: C/D layout = VGPR r, lane l -> row = r + 8*(l>>4), col = l&15
    const int ehalf = lane >> 4;
    const int ecol  = lane & 15;
    #pragma unroll
    for (int tm = 0; tm < 2; ++tm) {
        #pragma unroll
        for (int tn = 0; tn < 2; ++tn) {
            const int r0 = rowBase + wm * 32 + tm * 16 + ehalf * 8;
            const int c  = colBase + wn * 32 + tn * 16 + ecol;
            const v8f mu = accMu[tm][tn];
            const v8f va = accVar[tm][tn];
            #pragma unroll
            for (int r = 0; r < 8; ++r) {
                const size_t row = (size_t)(r0 + r);
                if (EPI == 0) {
                    muOut[row * 512 + c]  = mu[r];
                    varOut[row * 512 + c] = va[r];
                } else {
                    const float sig = sqrtf(va[r] + 1e-12f);
                    float x = mu[r] + sig * eps[row * 512 + c];
                    x = x > 0.f ? x : 0.01f * x;             // leaky_relu
                    hOut[row * 544 + c] = (_Float16)x;
                }
            }
        }
    }
}

// ---------------------------------------------------------------------------
// Conversion / init kernels
// ---------------------------------------------------------------------------
__global__ void convert_x(const float* __restrict__ x, _Float16* __restrict__ X)
{
    int idx = blockIdx.x * 256 + threadIdx.x;                 // 2048*800
    if (idx >= 2048 * 800) return;
    int b = idx / 800, k = idx - b * 800;
    X[idx] = (k < 784) ? (_Float16)x[b * 784 + k] : (_Float16)0.0f;
}

// mean/scale: [Ksrc x Nsrc] row-major. dst Wt/Vt: [Ndst x Kpad] (transposed, zero padded).
__global__ void convert_w(const float* __restrict__ mean, const float* __restrict__ scale,
                          const float* __restrict__ drop,
                          _Float16* __restrict__ Wt, _Float16* __restrict__ Vt,
                          int Ksrc, int Kpad, int Nsrc, int Ndst)
{
    int idx = blockIdx.x * 256 + threadIdx.x;
    if (idx >= Ndst * Kpad) return;
    int n = idx / Kpad, k = idx - n * Kpad;
    _Float16 w = (_Float16)0.0f, v = (_Float16)0.0f;
    if (k < Ksrc && n < Nsrc) {
        float d = drop ? *drop : 1.0f;
        float s = d * scale[(size_t)k * Nsrc + n];
        w = (_Float16)mean[(size_t)k * Nsrc + n];
        v = (_Float16)(s * s);
    }
    Wt[idx] = w;
    Vt[idx] = v;
}

// layer-1 activation (broadcast mu1/var1 over S samples) + ones/zero tail
__global__ void act1_kernel(const float* __restrict__ mu1, const float* __restrict__ var1,
                            const float* __restrict__ eps1, _Float16* __restrict__ h)
{
    const int g = blockIdx.x;             // 0..20479 (= s*2048 + b)
    const int b = g & 2047;
    for (int col = threadIdx.x; col < 544; col += 256) {
        _Float16 o;
        if (col < 512) {
            const float sig = sqrtf(var1[(size_t)b * 512 + col] + 1e-12f);
            float x = mu1[(size_t)b * 512 + col] + sig * eps1[(size_t)g * 512 + col];
            x = x > 0.f ? x : 0.01f * x;
            o = (_Float16)x;
        } else {
            o = (col == 512) ? (_Float16)1.0f : (_Float16)0.0f;
        }
        h[(size_t)g * 544 + col] = o;
    }
}

// set ones column + zero K-pad for a ping-pong h buffer (epilogue only writes cols 0..511)
__global__ void init_tail(_Float16* __restrict__ h)
{
    int idx = blockIdx.x * 256 + threadIdx.x;                 // 20480*32
    if (idx >= 20480 * 32) return;
    int g = idx >> 5, c = 512 + (idx & 31);
    h[(size_t)g * 544 + c] = (c == 512) ? (_Float16)1.0f : (_Float16)0.0f;
}

// ---------------------------------------------------------------------------
// Layer 4: tiny [K=544 -> C=10] dual-dot + fused log_softmax. Weights broadcast
// from LDS (all lanes read same address -> conflict-free LDS broadcast).
// ---------------------------------------------------------------------------
__global__ __launch_bounds__(256)
void layer4_kernel(const _Float16* __restrict__ h,
                   const _Float16* __restrict__ W4t, const _Float16* __restrict__ V4t,
                   const float* __restrict__ eps4, float* __restrict__ out)
{
    __shared__ _Float16 sW[16 * 544];
    __shared__ _Float16 sV[16 * 544];
    for (int i = threadIdx.x; i < (16 * 544) / 8; i += 256) {
        ((uint4*)sW)[i] = ((const uint4*)W4t)[i];
        ((uint4*)sV)[i] = ((const uint4*)V4t)[i];
    }
    __syncthreads();

    const int g = blockIdx.x * 256 + threadIdx.x;             // 0..20479
    float mu[10], var[10];
    #pragma unroll
    for (int c = 0; c < 10; ++c) { mu[c] = 0.f; var[c] = 0.f; }

    for (int k = 0; k < 544; k += 8) {
        F16x8 u;
        u.q = *(const uint4*)(h + (size_t)g * 544 + k);
        #pragma unroll
        for (int j = 0; j < 8; ++j) {
            const float xf = (float)u.x[j];
            const float xs = xf * xf;
            #pragma unroll
            for (int c = 0; c < 10; ++c) {
                mu[c]  += xf * (float)sW[c * 544 + k + j];
                var[c] += xs * (float)sV[c * 544 + k + j];
            }
        }
    }

    float l[10], m = -3.4e38f;
    #pragma unroll
    for (int c = 0; c < 10; ++c) {
        const float e = eps4[(size_t)g * 10 + c];
        l[c] = mu[c] + sqrtf(var[c] + 1e-12f) * e;
        m = fmaxf(m, l[c]);
    }
    float s = 0.f;
    #pragma unroll
    for (int c = 0; c < 10; ++c) s += __expf(l[c] - m);
    const float lse = __logf(s) + m;
    #pragma unroll
    for (int c = 0; c < 10; ++c) out[(size_t)g * 10 + c] = l[c] - lse;
}

// ---------------------------------------------------------------------------
extern "C" void kernel_launch(void* const* d_in, const int* in_sizes, int n_in,
                              void* d_out, int out_size, void* d_ws, size_t ws_size,
                              hipStream_t stream)
{
    const float* inputs   = (const float*)d_in[0];
    const float* a1_mean  = (const float*)d_in[1];
    const float* a1_scale = (const float*)d_in[2];
    const float* a1_drop  = (const float*)d_in[3];
    const float* a2_mean  = (const float*)d_in[4];
    const float* a2_scale = (const float*)d_in[5];
    const float* a2_drop  = (const float*)d_in[6];
    const float* a3_mean  = (const float*)d_in[7];
    const float* a3_scale = (const float*)d_in[8];
    const float* a3_drop  = (const float*)d_in[9];
    const float* a4_mean  = (const float*)d_in[10];
    const float* a4_scale = (const float*)d_in[11];
    const float* eps1     = (const float*)d_in[12];
    const float* eps2     = (const float*)d_in[13];
    const float* eps3     = (const float*)d_in[14];
    const float* eps4     = (const float*)d_in[15];

    char* ws = (char*)d_ws;
    size_t off = 0;
    auto alloc = [&](size_t bytes) -> void* {
        void* p = ws + off;
        off += (bytes + 255) & ~(size_t)255;
        return p;
    };

    _Float16* Xf16 = (_Float16*)alloc(2048ull * 800 * 2);
    _Float16* W1t  = (_Float16*)alloc(512ull * 800 * 2);
    _Float16* V1t  = (_Float16*)alloc(512ull * 800 * 2);
    _Float16* W2t  = (_Float16*)alloc(512ull * 544 * 2);
    _Float16* V2t  = (_Float16*)alloc(512ull * 544 * 2);
    _Float16* W3t  = (_Float16*)alloc(512ull * 544 * 2);
    _Float16* V3t  = (_Float16*)alloc(512ull * 544 * 2);
    _Float16* W4t  = (_Float16*)alloc(16ull  * 544 * 2);
    _Float16* V4t  = (_Float16*)alloc(16ull  * 544 * 2);
    float*    mu1  = (float*)   alloc(2048ull * 512 * 4);
    float*    var1 = (float*)   alloc(2048ull * 512 * 4);
    _Float16* hA   = (_Float16*)alloc(20480ull * 544 * 2);
    _Float16* hB   = (_Float16*)alloc(20480ull * 544 * 2);

    // --- conversions (weights transposed to N-major, K padded to mult. of 32) ---
    convert_x<<<(2048 * 800 + 255) / 256, 256, 0, stream>>>(inputs, Xf16);
    convert_w<<<(512 * 800 + 255) / 256, 256, 0, stream>>>(a1_mean, a1_scale, a1_drop,
                                                           W1t, V1t, 784, 800, 512, 512);
    convert_w<<<(512 * 544 + 255) / 256, 256, 0, stream>>>(a2_mean, a2_scale, a2_drop,
                                                           W2t, V2t, 513, 544, 512, 512);
    convert_w<<<(512 * 544 + 255) / 256, 256, 0, stream>>>(a3_mean, a3_scale, a3_drop,
                                                           W3t, V3t, 513, 544, 512, 512);
    convert_w<<<(16 * 544 + 255) / 256, 256, 0, stream>>>(a4_mean, a4_scale, nullptr,
                                                          W4t, V4t, 513, 544, 10, 16);
    init_tail<<<(20480 * 32 + 255) / 256, 256, 0, stream>>>(hB);

    // --- layer 1 (sample-independent): mu1/var1 once, M=2048 ---
    gemm_fused<0><<<dim3(2048 / BM, 512 / BN), 256, 0, stream>>>(
        Xf16, W1t, V1t, 800, nullptr, mu1, var1, nullptr);

    // --- broadcast over S samples + activation -> hA [20480 x 544] ---
    act1_kernel<<<20480, 256, 0, stream>>>(mu1, var1, eps1, hA);

    // --- layers 2 & 3 batched over all samples (M = 20480) ---
    gemm_fused<1><<<dim3(20480 / BM, 512 / BN), 256, 0, stream>>>(
        hA, W2t, V2t, 544, eps2, nullptr, nullptr, hB);
    gemm_fused<1><<<dim3(20480 / BM, 512 / BN), 256, 0, stream>>>(
        hB, W3t, V3t, 544, eps3, nullptr, nullptr, hA);

    // --- layer 4 + log_softmax -> d_out [S*B, 10] f32 ---
    layer4_kernel<<<20480 / 256, 256, 0, stream>>>(hA, W4t, V4t, eps4, (float*)d_out);
}
